// GRU_42734924595877
// MI455X (gfx1250) — matrix operand order repaired and checked by reading, not compiled, not run on hardware
//
#include <hip/hip_runtime.h>

// GRU on MI455X (gfx1250): bf16 WMMA (v_wmma_f32_16x16x32_bf16) with fp32
// accumulate. Weights converted to bf16 once (12 MB -> L2 resident across all
// 512 timesteps). Two kernels per timestep give the cross-WGP sync the
// recurrence needs; stream ordering / graph capture handles dependencies.
//
// Round-2 tuning: split x/h K-loops (no per-iter cndmask addressing), dual
// accumulators to break the WMMA C->D dependency chain, and v_rcp_f32-based
// sigmoid/tanh instead of IEEE-divide expansions.

typedef __attribute__((ext_vector_type(16))) __bf16 v16bf;
typedef __attribute__((ext_vector_type(8)))  __bf16 v8bf;
typedef __attribute__((ext_vector_type(4)))  __bf16 v4bf;
typedef __attribute__((ext_vector_type(8)))  float  v8f;

constexpr int kB = 64, kS = 512, kI = 1024, kH = 1024, kIH = 2048, kO = 1024;

// sigmoid(x) = rcp(1 + e^-x); tanh(x) = 1 - 2*rcp(1 + e^{2x}).
// v_rcp_f32 (~1 ulp) is plenty for a bf16-precision recurrence.
__device__ __forceinline__ float fast_sigmoid(float x) {
  return __builtin_amdgcn_rcpf(1.0f + __expf(-x));
}
__device__ __forceinline__ float fast_tanh(float x) {
  return 1.0f - 2.0f * __builtin_amdgcn_rcpf(1.0f + __expf(2.0f * x));
}

// ---------------- prep kernels ----------------
__global__ __launch_bounds__(256)
void f32_to_bf16_v4(const float4* __restrict__ s, v4bf* __restrict__ d, int n4) {
  int i = blockIdx.x * blockDim.x + threadIdx.x;
  if (i < n4) {
    float4 v = s[i];
    v4bf o;
    o[0] = (__bf16)v.x; o[1] = (__bf16)v.y; o[2] = (__bf16)v.z; o[3] = (__bf16)v.w;
    d[i] = o;
  }
}

__global__ __launch_bounds__(256)
void init_h(float* __restrict__ hf, __bf16* __restrict__ hb, int n) {
  int i = blockIdx.x * blockDim.x + threadIdx.x;
  if (i < n) { hf[i] = 0.0f; hb[i] = (__bf16)0.0f; }
}

// ---------------- fragment loaders (wave32, contiguous k order) ----------------
// A-matrix 16x32 bf16, k order used here: lane row = lane&15; lanes<16: K[0..15];
// lanes>=16: K[16..31].  `base` points at K=kb of this row.
__device__ __forceinline__ v16bf load_a_frag(const __bf16* __restrict__ base, int half) {
  return *(const v16bf*)(base + half * 16);
}
// B-matrix 32x16 bf16, same k order as A: lane column = lane&15; lanes<16 hold K[0..15],
// lanes>=16 hold K[16..31] -> one contiguous 32B load from W[col][kb + 16*half].
__device__ __forceinline__ v16bf load_b_frag(const __bf16* __restrict__ wrow_kb, int half) {
  return *(const v16bf*)(wrow_kb + half * 16);
}

// 1024-long dot-strip with two independent accumulator chains.
__device__ __forceinline__ void mma_k1024(const __bf16* __restrict__ arow,
                                          const __bf16* __restrict__ wrow,
                                          int half, v8f& acc0, v8f& acc1) {
  #pragma unroll 2
  for (int kb = 0; kb < kI; kb += 64) {
    v16bf a0 = load_a_frag(arow + kb, half);
    v16bf b0 = load_b_frag(wrow + kb, half);
    acc0 = __builtin_amdgcn_wmma_f32_16x16x32_bf16(false, a0, false, b0,
                                                   (short)0, acc0, false, false);
    v16bf a1 = load_a_frag(arow + kb + 32, half);
    v16bf b1 = load_b_frag(wrow + kb + 32, half);
    acc1 = __builtin_amdgcn_wmma_f32_16x16x32_bf16(false, a1, false, b1,
                                                   (short)0, acc1, false, false);
  }
}

// ---------------- phase A: r and z gates ----------------
// 512 waves: 4 row tiles (B=64) x 128 col tiles (2048 gate cols: [Wr | Wz]).
__global__ __launch_bounds__(256)
void gru_gates_rz(const __bf16* __restrict__ xb, const __bf16* __restrict__ hb,
                  const float* __restrict__ hf,
                  const __bf16* __restrict__ Wr, const __bf16* __restrict__ Wz,
                  const float* __restrict__ br, const float* __restrict__ bz,
                  __bf16* __restrict__ rh, float* __restrict__ zf, int t) {
  const int lane = threadIdx.x & 31;
  const int wave = blockIdx.x * (blockDim.x >> 5) + (threadIdx.x >> 5);
  const int colTile = wave & 127;
  const int rowTile = wave >> 7;
  const int n0g = colTile * 16;
  const int m0  = rowTile * 16;
  const bool isZ = (n0g >= kH);
  const __bf16* __restrict__ W = isZ ? Wz : Wr;
  const int nloc = n0g - (isZ ? kH : 0);

  const int half = lane >> 4;
  const int arow = m0 + (lane & 15);                 // batch row for A loads
  const int col  = nloc + (lane & 15);               // gate-local output column

  const __bf16* __restrict__ ax = xb + ((size_t)arow * kS + t) * kI;
  const __bf16* __restrict__ ah = hb + (size_t)arow * kH;
  const __bf16* __restrict__ wrow = W + (size_t)col * kIH;

  v8f acc0 = {}, acc1 = {};
  mma_k1024(ax, wrow,       half, acc0, acc1);       // x   @ W[:, :1024]^T
  mma_k1024(ah, wrow + kI,  half, acc0, acc1);       // h   @ W[:, 1024:]^T

  const float bias = isZ ? bz[col] : br[col];
  #pragma unroll
  for (int i = 0; i < 8; ++i) {
    const int r = m0 + i + half * 8;                 // output row of acc[i]
    const size_t idx = (size_t)r * kH + col;
    const float g = fast_sigmoid(acc0[i] + acc1[i] + bias);
    if (isZ) zf[idx] = g;
    else     rh[idx] = (__bf16)(g * hf[idx]);        // r * h (bf16 for phase B)
  }
}

// ---------------- phase B: candidate + state update ----------------
// 256 waves: 4 row tiles x 64 col tiles (H=1024).
__global__ __launch_bounds__(256)
void gru_cand_update(const __bf16* __restrict__ xb, const __bf16* __restrict__ rh,
                     const float* __restrict__ zf,
                     const __bf16* __restrict__ Wt, const float* __restrict__ bt,
                     float* __restrict__ hf, __bf16* __restrict__ hb, int t) {
  const int lane = threadIdx.x & 31;
  const int wave = blockIdx.x * (blockDim.x >> 5) + (threadIdx.x >> 5);
  const int colTile = wave & 63;
  const int rowTile = wave >> 6;
  const int n0 = colTile * 16;
  const int m0 = rowTile * 16;

  const int half = lane >> 4;
  const int arow = m0 + (lane & 15);
  const int col  = n0 + (lane & 15);

  const __bf16* __restrict__ ax = xb + ((size_t)arow * kS + t) * kI;
  const __bf16* __restrict__ ar = rh + (size_t)arow * kH;
  const __bf16* __restrict__ wrow = Wt + (size_t)col * kIH;

  v8f acc0 = {}, acc1 = {};
  mma_k1024(ax, wrow,      half, acc0, acc1);        // x    @ Wt[:, :1024]^T
  mma_k1024(ar, wrow + kI, half, acc0, acc1);        // r*h  @ Wt[:, 1024:]^T

  const float bias = bt[col];
  #pragma unroll
  for (int i = 0; i < 8; ++i) {
    const int r = m0 + i + half * 8;
    const size_t idx = (size_t)r * kH + col;
    const float ht = fast_tanh(acc0[i] + acc1[i] + bias);
    const float z  = zf[idx];
    const float hn = z * hf[idx] + (1.0f - z) * ht;
    hf[idx] = hn;                                    // fp32 state (1 lane per elem)
    hb[idx] = (__bf16)hn;                            // bf16 copy for next step
  }
}

// ---------------- final projection: out = h @ Wo^T + bo ----------------
__global__ __launch_bounds__(256)
void gru_out(const __bf16* __restrict__ hb, const __bf16* __restrict__ Wo,
             const float* __restrict__ bo, float* __restrict__ out) {
  const int lane = threadIdx.x & 31;
  const int wave = blockIdx.x * (blockDim.x >> 5) + (threadIdx.x >> 5);
  const int colTile = wave & 63;
  const int rowTile = wave >> 6;
  const int n0 = colTile * 16;
  const int m0 = rowTile * 16;

  const int half = lane >> 4;
  const int arow = m0 + (lane & 15);
  const int col  = n0 + (lane & 15);

  const __bf16* __restrict__ ah   = hb + (size_t)arow * kH;
  const __bf16* __restrict__ wrow = Wo + (size_t)col * kH;

  v8f acc0 = {}, acc1 = {};
  mma_k1024(ah, wrow, half, acc0, acc1);

  const float bias = bo[col];
  #pragma unroll
  for (int i = 0; i < 8; ++i) {
    const int r = m0 + i + half * 8;
    out[(size_t)r * kO + col] = acc0[i] + acc1[i] + bias;
  }
}

__global__ __launch_bounds__(256)
void copy_h(const float* __restrict__ hf, float* __restrict__ dst, int n) {
  int i = blockIdx.x * blockDim.x + threadIdx.x;
  if (i < n) dst[i] = hf[i];
}

// ---------------- host side ----------------
extern "C" void kernel_launch(void* const* d_in, const int* in_sizes, int n_in,
                              void* d_out, int out_size, void* d_ws, size_t ws_size,
                              hipStream_t stream) {
  (void)in_sizes; (void)n_in; (void)out_size; (void)ws_size;
  const float* x  = (const float*)d_in[0];
  const float* Wr = (const float*)d_in[1];
  const float* br = (const float*)d_in[2];
  const float* Wz = (const float*)d_in[3];
  const float* bz = (const float*)d_in[4];
  const float* Wt = (const float*)d_in[5];
  const float* bt = (const float*)d_in[6];
  const float* Wo = (const float*)d_in[7];
  const float* bo = (const float*)d_in[8];
  float* out = (float*)d_out;

  // Workspace carve-up (~79 MB total).
  char* p = (char*)d_ws;
  __bf16* Wr_b = (__bf16*)p; p += (size_t)kH * kIH * 2;
  __bf16* Wz_b = (__bf16*)p; p += (size_t)kH * kIH * 2;
  __bf16* Wt_b = (__bf16*)p; p += (size_t)kH * kIH * 2;
  __bf16* Wo_b = (__bf16*)p; p += (size_t)kO * kH * 2;
  __bf16* x_b  = (__bf16*)p; p += (size_t)kB * kS * kI * 2;
  float*  h_f  = (float*)p;  p += (size_t)kB * kH * 4;
  __bf16* h_b  = (__bf16*)p; p += (size_t)kB * kH * 2;
  __bf16* rh_b = (__bf16*)p; p += (size_t)kB * kH * 2;
  float*  z_f  = (float*)p;  p += (size_t)kB * kH * 4;

  auto conv = [&](const float* src, __bf16* dst, size_t n) {
    int n4 = (int)(n / 4);
    f32_to_bf16_v4<<<(n4 + 255) / 256, 256, 0, stream>>>((const float4*)src, (v4bf*)dst, n4);
  };
  conv(Wr, Wr_b, (size_t)kH * kIH);
  conv(Wz, Wz_b, (size_t)kH * kIH);
  conv(Wt, Wt_b, (size_t)kH * kIH);
  conv(Wo, Wo_b, (size_t)kO * kH);
  conv(x,  x_b,  (size_t)kB * kS * kI);
  init_h<<<(kB * kH + 255) / 256, 256, 0, stream>>>(h_f, h_b, kB * kH);

  // Recurrence: two kernels per step; stream order provides the global sync.
  for (int t = 0; t < kS; ++t) {
    gru_gates_rz<<<64, 256, 0, stream>>>(x_b, h_b, h_f, Wr_b, Wz_b, br, bz, rh_b, z_f, t);
    gru_cand_update<<<32, 256, 0, stream>>>(x_b, rh_b, z_f, Wt_b, bt, h_f, h_b, t);
  }

  gru_out<<<32, 256, 0, stream>>>(h_b, Wo_b, bo, out);
  copy_h<<<(kB * kH + 255) / 256, 256, 0, stream>>>(h_f, out + (size_t)kB * kO, kB * kH);
}
